// AttentionLayer_34857954574789
// MI455X (gfx1250) — matrix operand order repaired and checked by reading, not compiled, not run on hardware
//
#include <hip/hip_runtime.h>

typedef __attribute__((ext_vector_type(16))) __bf16 v16bf;
typedef __attribute__((ext_vector_type(8)))  float  v8f;
typedef unsigned int   u32;
typedef unsigned short u16;

#define B_   2
#define S_   2048
#define D_   2048
#define H_   16
#define KVH_ 8
#define HD_  128

// LDS pitches in dwords — 16-byte aligned rows so fragment reads lower to
// ds_load_b128, and gcd(pitch,64) small enough that 16 lanes hit 16 banks.
#define GP_  20   // GEMM A / B^T row pitch (80 B)
#define VP_  36   // attention V^T / P row pitch (144 B)

__device__ __forceinline__ u16 f2bf(float f) {
    u32 u = __float_as_uint(f);
    u32 r = u + 0x7FFFu + ((u >> 16) & 1u);   // round-to-nearest-even
    return (u16)(r >> 16);
}

union Frag { v16bf v; u32 u[8]; };
union Acc  { v8f   v; float f[8]; };

// ---------------------------------------------------------------- convert
__global__ void cvt_f32_bf16_kernel(const float* __restrict__ in,
                                    u16* __restrict__ out, int n) {
    int i = blockIdx.x * blockDim.x + threadIdx.x;
    int stride = gridDim.x * blockDim.x;
    for (; i < n; i += stride) out[i] = f2bf(in[i]);
}

// ---------------------------------------------------------------- QKV + RoPE
// x @ [wq|wk|wv] in 128x128 tiles; RoPE fused for q,k tiles.
// grid = (32 M-tiles, 32 N-tiles), block = 256 (8 waves).
__global__ __launch_bounds__(256) void qkv_rope_kernel(
    const u16* __restrict__ Xb,
    const u16* __restrict__ Wqb, const u16* __restrict__ Wkb,
    const u16* __restrict__ Wvb,
    const float* __restrict__ cosT, const float* __restrict__ sinT,
    u16* __restrict__ Qb, u16* __restrict__ Kb, u16* __restrict__ Vb)
{
    __shared__ u32 ldsA[2][128 * GP_];
    __shared__ u32 ldsBT[2][128 * GP_];

    const int tid  = threadIdx.x;
    const int wave = tid >> 5, lane = tid & 31;
    const int l16  = lane & 15, half = lane >> 4;
    const int mt = blockIdx.x;
    const int nt = blockIdx.y;           // 0..15 q, 16..23 k, 24..31 v

    const u32* Xu = (const u32*)Xb;
    const u32* Wu; int wpitch_u, ncol0_u;
    if (nt < 16)      { Wu = (const u32*)Wqb; wpitch_u = 1024; ncol0_u = nt * 64; }
    else if (nt < 24) { Wu = (const u32*)Wkb; wpitch_u = 512;  ncol0_u = (nt - 16) * 64; }
    else              { Wu = (const u32*)Wvb; wpitch_u = 512;  ncol0_u = (nt - 24) * 64; }

    const int arow  = tid >> 1;        // 0..127
    const int aucol = (tid & 1) * 8;   // 0 or 8
    const int brow  = tid >> 3;        // 0..31
    const int bucol = (tid & 7) * 8;   // 0..56
    const long mbase = (long)mt * 128;

    Acc acc[8];
    #pragma unroll
    for (int t = 0; t < 8; ++t)
        #pragma unroll
        for (int i = 0; i < 8; ++i) acc[t].f[i] = 0.f;

    u32 aReg[8], bReg[8];
    #pragma unroll
    for (int j = 0; j < 8; ++j) {
        aReg[j] = Xu[(mbase + arow) * 1024 + aucol + j];
        bReg[j] = Wu[(long)brow * wpitch_u + ncol0_u + bucol + j];
    }

    for (int ks = 0; ks < 64; ++ks) {
        const int cur = ks & 1;
        u32* A  = ldsA[cur];
        u32* BT = ldsBT[cur];
        #pragma unroll
        for (int j = 0; j < 8; ++j)
            A[arow * GP_ + aucol + j] = aReg[j];
        #pragma unroll
        for (int j = 0; j < 8; ++j) {
            int c = (bucol + j) * 2;
            u16* bt = (u16*)BT;
            bt[(c + 0) * (2 * GP_) + brow] = (u16)(bReg[j] & 0xFFFFu);
            bt[(c + 1) * (2 * GP_) + brow] = (u16)(bReg[j] >> 16);
        }
        __syncthreads();
        if (ks + 1 < 64) {   // stage next tile into registers (overlaps WMMA)
            #pragma unroll
            for (int j = 0; j < 8; ++j) {
                aReg[j] = Xu[(mbase + arow) * 1024 + (ks + 1) * 16 + aucol + j];
                bReg[j] = Wu[(long)((ks + 1) * 32 + brow) * wpitch_u + ncol0_u + bucol + j];
            }
            if (ks + 2 < 64) {
                __builtin_prefetch(&Xu[(mbase + arow) * 1024 + (ks + 2) * 16 + aucol], 0, 3);
                __builtin_prefetch(&Wu[(long)((ks + 2) * 32 + brow) * wpitch_u + ncol0_u + bucol], 0, 3);
            }
        }
        Frag a;
        const u32* arp = &A[(wave * 16 + l16) * GP_];
        #pragma unroll
        for (int j = 0; j < 4; ++j) {
            a.u[j]     = arp[half * 8 + j];
            a.u[4 + j] = arp[half * 8 + 4 + j];
        }
        #pragma unroll
        for (int g = 0; g < 2; ++g) {
            Frag bfr[4];
            #pragma unroll
            for (int t = 0; t < 4; ++t) {
                const u32* brp = &BT[((g * 4 + t) * 16 + l16) * GP_];
                #pragma unroll
                for (int j = 0; j < 8; ++j) bfr[t].u[j] = brp[half * 8 + j];
            }
            #pragma unroll
            for (int t = 0; t < 4; ++t)
                acc[g * 4 + t].v = __builtin_amdgcn_wmma_f32_16x16x32_bf16(
                    false, a.v, false, bfr[t].v, (short)0, acc[g * 4 + t].v,
                    false, false);
        }
    }

    // epilogue: RoPE (q,k) and scatter to [b][h][s][hd] bf16
    #pragma unroll
    for (int j = 0; j < 8; ++j) {
        int m = mt * 128 + wave * 16 + half * 8 + j;
        int b = m >> 11;
        int s = m & 2047;
        if (nt < 24) {
            const float* cr = cosT + s * HD_;
            const float* sr = sinT + s * HD_;
            float vals[8];
            #pragma unroll
            for (int t = 0; t < 8; ++t) {
                int hd   = t * 16 + l16;
                float x0 = acc[t].f[j];
                float xp = (t < 4) ? acc[t + 4].f[j] : acc[t - 4].f[j];
                float sg = (t < 4) ? -1.f : 1.f;
                vals[t] = x0 * cr[hd] + sg * xp * sr[hd];
            }
            u16* dst = (nt < 16)
                ? Qb + (((long)b * H_   + nt)        * S_ + s) * HD_
                : Kb + (((long)b * KVH_ + (nt - 16)) * S_ + s) * HD_;
            #pragma unroll
            for (int t = 0; t < 8; ++t) dst[t * 16 + l16] = f2bf(vals[t]);
        } else {
            u16* dst = Vb + (((long)b * KVH_ + (nt - 24)) * S_ + s) * HD_;
            #pragma unroll
            for (int t = 0; t < 8; ++t) dst[t * 16 + l16] = f2bf(acc[t].f[j]);
        }
    }
}

// ---------------------------------------------------------------- attention
// Flash attention: block per (b, h, 128-query tile); 64-key blocks.
__global__ __launch_bounds__(256) void attn_kernel(
    const u16* __restrict__ Qb, const u16* __restrict__ Kb,
    const u16* __restrict__ Vb, u16* __restrict__ Ab)
{
    __shared__ u32 ldsVT[2][128 * VP_];    // V^T: [hd=128][kp=64] halves
    __shared__ u32 ldsP[8 * 16 * VP_];     // per-wave P: 16 q x 64 kp

    const int tid  = threadIdx.x;
    const int wave = tid >> 5, lane = tid & 31;
    const int l16  = lane & 15, half = lane >> 4;
    const int qt = blockIdx.x;
    const int h  = blockIdx.y;
    const int b  = blockIdx.z;
    const int kvh = h >> 1;               // N_REP = 2

    const u16* Qh = Qb + ((long)b * H_   + h)   * S_ * HD_;
    const u16* Kh = Kb + ((long)b * KVH_ + kvh) * S_ * HD_;
    const u16* Vh = Vb + ((long)b * KVH_ + kvh) * S_ * HD_;

    // preload Q fragments (4 head-dim K-steps) — reused for every key block
    Frag qf[4];
    {
        const u32* qrow = (const u32*)(Qh + (long)(qt * 128 + wave * 16 + l16) * HD_);
        #pragma unroll
        for (int s = 0; s < 4; ++s)
            #pragma unroll
            for (int j = 0; j < 4; ++j) {
                qf[s].u[j]     = qrow[s * 16 + half * 8 + j];
                qf[s].u[4 + j] = qrow[s * 16 + half * 8 + 4 + j];
            }
    }

    Acc o[8];
    #pragma unroll
    for (int t = 0; t < 8; ++t)
        #pragma unroll
        for (int i = 0; i < 8; ++i) o[t].f[i] = 0.f;
    float mrow[8], lrow[8];
    #pragma unroll
    for (int j = 0; j < 8; ++j) { mrow[j] = -3.0e38f; lrow[j] = 0.f; }

    const float scale = 0.088388347648318447f;   // 1/sqrt(128)
    u16*       Ph = (u16*)(ldsP + wave * 16 * VP_);
    const u32* Pu = ldsP + wave * 16 * VP_;

    const int kp = tid >> 2;              // 0..63
    const int c0 = (tid & 3) * 32;        // hd column base
    u32 vreg[16];
    #pragma unroll
    for (int j = 0; j < 16; ++j)
        vreg[j] = ((const u32*)(Vh + (long)kp * HD_))[(c0 >> 1) + j];

    const int nkb = 2 * qt + 2;           // causal: 64-key blocks
    for (int kb = 0; kb < nkb; ++kb) {
        u16* vt = (u16*)ldsVT[kb & 1];
        #pragma unroll
        for (int j = 0; j < 16; ++j) {
            int c = c0 + j * 2;
            vt[(c + 0) * (2 * VP_) + kp] = (u16)(vreg[j] & 0xFFFFu);
            vt[(c + 1) * (2 * VP_) + kp] = (u16)(vreg[j] >> 16);
        }
        __syncthreads();
        if (kb + 1 < nkb) {   // pipeline next V block into registers
            const u32* vrow = (const u32*)(Vh + (long)((kb + 1) * 64 + kp) * HD_);
            #pragma unroll
            for (int j = 0; j < 16; ++j) vreg[j] = vrow[(c0 >> 1) + j];
        }

        // S = Q @ K^T (K fragments batched per hd-step)
        Acc sa[4];
        #pragma unroll
        for (int t = 0; t < 4; ++t)
            #pragma unroll
            for (int i = 0; i < 8; ++i) sa[t].f[i] = 0.f;
        #pragma unroll
        for (int s = 0; s < 4; ++s) {
            Frag bk[4];
            #pragma unroll
            for (int t = 0; t < 4; ++t) {
                const u32* krow =
                    (const u32*)(Kh + (long)(kb * 64 + t * 16 + l16) * HD_);
                #pragma unroll
                for (int j = 0; j < 8; ++j) bk[t].u[j] = krow[s * 16 + half * 8 + j];
            }
            #pragma unroll
            for (int t = 0; t < 4; ++t)
                sa[t].v = __builtin_amdgcn_wmma_f32_16x16x32_bf16(
                    false, qf[s].v, false, bk[t].v, (short)0, sa[t].v, false, false);
        }

        // mask + online softmax (lane-parallel rows via shfl_xor within halves)
        const bool diag = (kb >= 2 * qt);
        #pragma unroll
        for (int j = 0; j < 8; ++j) {
            int q_g = qt * 128 + wave * 16 + half * 8 + j;
            float mx = -3.0e38f;
            #pragma unroll
            for (int t = 0; t < 4; ++t) {
                float v = sa[t].f[j] * scale;
                if (diag) {
                    int k_g = kb * 64 + t * 16 + l16;
                    v += (k_g > q_g) ? -1.0e9f : 0.f;
                }
                sa[t].f[j] = v;
                mx = fmaxf(mx, v);
            }
            #pragma unroll
            for (int mk = 1; mk < 16; mk <<= 1)
                mx = fmaxf(mx, __shfl_xor(mx, mk, 32));
            float mnew  = fmaxf(mrow[j], mx);
            float alpha = __expf(mrow[j] - mnew);
            mrow[j] = mnew;
            float rsum = 0.f;
            #pragma unroll
            for (int t = 0; t < 4; ++t) {
                float p = __expf(sa[t].f[j] - mnew);
                rsum += p;
                Ph[(half * 8 + j) * (2 * VP_) + t * 16 + l16] = f2bf(p);
            }
            #pragma unroll
            for (int mk = 1; mk < 16; mk <<= 1)
                rsum += __shfl_xor(rsum, mk, 32);
            lrow[j] = lrow[j] * alpha + rsum;
            #pragma unroll
            for (int t = 0; t < 8; ++t) o[t].f[j] *= alpha;
        }

        // O += P @ V  (P wave-private LDS; V^T shared LDS)
        const u32* VT = ldsVT[kb & 1];
        #pragma unroll
        for (int s = 0; s < 2; ++s) {
            Frag pa;
            const u32* prow = Pu + l16 * VP_;
            #pragma unroll
            for (int j = 0; j < 4; ++j) {
                pa.u[j]     = prow[s * 16 + half * 8 + j];
                pa.u[4 + j] = prow[s * 16 + half * 8 + 4 + j];
            }
            #pragma unroll
            for (int g = 0; g < 2; ++g) {
                Frag vbf[4];
                #pragma unroll
                for (int t = 0; t < 4; ++t) {
                    const u32* vtr = &VT[((g * 4 + t) * 16 + l16) * VP_];
                    #pragma unroll
                    for (int j = 0; j < 8; ++j)
                        vbf[t].u[j] = vtr[s * 16 + half * 8 + j];
                }
                #pragma unroll
                for (int t = 0; t < 4; ++t)
                    o[g * 4 + t].v = __builtin_amdgcn_wmma_f32_16x16x32_bf16(
                        false, pa.v, false, vbf[t].v, (short)0,
                        o[g * 4 + t].v, false, false);
            }
        }
        // no trailing barrier: VT double-buffered, P wave-private
    }

    // normalize and store to attn buffer [b][s][h*128+hd] (bf16)
    #pragma unroll
    for (int j = 0; j < 8; ++j) {
        float inv = 1.0f / lrow[j];
        int srow = qt * 128 + wave * 16 + half * 8 + j;
        u16* dst = Ab + ((long)b * S_ + srow) * (long)(H_ * HD_) + h * HD_;
        #pragma unroll
        for (int t = 0; t < 8; ++t) dst[t * 16 + l16] = f2bf(o[t].f[j] * inv);
    }
}

// ---------------------------------------------------------------- out proj
__global__ __launch_bounds__(256) void out_proj_kernel(
    const u16* __restrict__ Ab, const u16* __restrict__ Wob,
    float* __restrict__ out)
{
    __shared__ u32 ldsA[2][128 * GP_];
    __shared__ u32 ldsBT[2][128 * GP_];

    const int tid  = threadIdx.x;
    const int wave = tid >> 5, lane = tid & 31;
    const int l16  = lane & 15, half = lane >> 4;
    const int mt = blockIdx.x;      // 0..31
    const int nt = blockIdx.y;      // 0..15

    const u32* Au = (const u32*)Ab;
    const u32* Wu = (const u32*)Wob;
    const int arow  = tid >> 1;
    const int aucol = (tid & 1) * 8;
    const int brow  = tid >> 3;
    const int bucol = (tid & 7) * 8;
    const long mbase = (long)mt * 128;
    const int ncol0_u = nt * 64;

    Acc acc[8];
    #pragma unroll
    for (int t = 0; t < 8; ++t)
        #pragma unroll
        for (int i = 0; i < 8; ++i) acc[t].f[i] = 0.f;

    u32 aReg[8], bReg[8];
    #pragma unroll
    for (int j = 0; j < 8; ++j) {
        aReg[j] = Au[(mbase + arow) * 1024 + aucol + j];
        bReg[j] = Wu[(long)brow * 1024 + ncol0_u + bucol + j];
    }

    for (int ks = 0; ks < 64; ++ks) {
        const int cur = ks & 1;
        u32* A  = ldsA[cur];
        u32* BT = ldsBT[cur];
        #pragma unroll
        for (int j = 0; j < 8; ++j)
            A[arow * GP_ + aucol + j] = aReg[j];
        #pragma unroll
        for (int j = 0; j < 8; ++j) {
            int c = (bucol + j) * 2;
            u16* bt = (u16*)BT;
            bt[(c + 0) * (2 * GP_) + brow] = (u16)(bReg[j] & 0xFFFFu);
            bt[(c + 1) * (2 * GP_) + brow] = (u16)(bReg[j] >> 16);
        }
        __syncthreads();
        if (ks + 1 < 64) {
            #pragma unroll
            for (int j = 0; j < 8; ++j) {
                aReg[j] = Au[(mbase + arow) * 1024 + (ks + 1) * 16 + aucol + j];
                bReg[j] = Wu[(long)((ks + 1) * 32 + brow) * 1024 + ncol0_u + bucol + j];
            }
            if (ks + 2 < 64) {
                __builtin_prefetch(&Au[(mbase + arow) * 1024 + (ks + 2) * 16 + aucol], 0, 3);
                __builtin_prefetch(&Wu[(long)((ks + 2) * 32 + brow) * 1024 + ncol0_u + bucol], 0, 3);
            }
        }
        Frag a;
        const u32* arp = &A[(wave * 16 + l16) * GP_];
        #pragma unroll
        for (int j = 0; j < 4; ++j) {
            a.u[j]     = arp[half * 8 + j];
            a.u[4 + j] = arp[half * 8 + 4 + j];
        }
        #pragma unroll
        for (int g = 0; g < 2; ++g) {
            Frag bfr[4];
            #pragma unroll
            for (int t = 0; t < 4; ++t) {
                const u32* brp = &BT[((g * 4 + t) * 16 + l16) * GP_];
                #pragma unroll
                for (int j = 0; j < 8; ++j) bfr[t].u[j] = brp[half * 8 + j];
            }
            #pragma unroll
            for (int t = 0; t < 4; ++t)
                acc[g * 4 + t].v = __builtin_amdgcn_wmma_f32_16x16x32_bf16(
                    false, a.v, false, bfr[t].v, (short)0, acc[g * 4 + t].v,
                    false, false);
        }
    }

    #pragma unroll
    for (int j = 0; j < 8; ++j) {
        long m = mt * 128 + wave * 16 + half * 8 + j;
        float* dst = out + m * 2048 + nt * 128;
        #pragma unroll
        for (int t = 0; t < 8; ++t) dst[t * 16 + l16] = acc[t].f[j];
    }
}

// ---------------------------------------------------------------- launcher
extern "C" void kernel_launch(void* const* d_in, const int* in_sizes, int n_in,
                              void* d_out, int out_size, void* d_ws, size_t ws_size,
                              hipStream_t stream) {
    (void)in_sizes; (void)n_in; (void)out_size; (void)ws_size;
    const float* x    = (const float*)d_in[0];
    const float* wq   = (const float*)d_in[1];
    const float* wk   = (const float*)d_in[2];
    const float* wv   = (const float*)d_in[3];
    const float* wo   = (const float*)d_in[4];
    const float* cosT = (const float*)d_in[5];
    const float* sinT = (const float*)d_in[6];
    // d_in[7] = mask: causal structure computed analytically in-kernel.
    float* out = (float*)d_out;

    u16* ws = (u16*)d_ws;
    size_t off = 0;
    u16* Xb  = ws + off; off += (size_t)B_ * S_ * D_;
    u16* Wqb = ws + off; off += (size_t)D_ * H_ * HD_;
    u16* Wkb = ws + off; off += (size_t)D_ * KVH_ * HD_;
    u16* Wvb = ws + off; off += (size_t)D_ * KVH_ * HD_;
    u16* Wob = ws + off; off += (size_t)H_ * HD_ * D_;
    u16* Qb  = ws + off; off += (size_t)B_ * H_ * S_ * HD_;
    u16* Kb  = ws + off; off += (size_t)B_ * KVH_ * S_ * HD_;
    u16* Vb  = ws + off; off += (size_t)B_ * KVH_ * S_ * HD_;
    u16* Ab  = ws + off; off += (size_t)B_ * S_ * H_ * HD_;

    cvt_f32_bf16_kernel<<<1024, 256, 0, stream>>>(x,  Xb,  B_ * S_ * D_);
    cvt_f32_bf16_kernel<<<1024, 256, 0, stream>>>(wq, Wqb, D_ * H_ * HD_);
    cvt_f32_bf16_kernel<<<1024, 256, 0, stream>>>(wk, Wkb, D_ * KVH_ * HD_);
    cvt_f32_bf16_kernel<<<1024, 256, 0, stream>>>(wv, Wvb, D_ * KVH_ * HD_);
    cvt_f32_bf16_kernel<<<1024, 256, 0, stream>>>(wo, Wob, H_ * HD_ * D_);

    qkv_rope_kernel<<<dim3(32, 32), 256, 0, stream>>>(
        Xb, Wqb, Wkb, Wvb, cosT, sinT, Qb, Kb, Vb);
    attn_kernel<<<dim3(16, 16, 2), 256, 0, stream>>>(Qb, Kb, Vb, Ab);
    out_proj_kernel<<<dim3(32, 16), 256, 0, stream>>>(Ab, Wob, out);
}
